// BotDCGCGraphAutoEncoder_50835232915952
// MI455X (gfx1250) — matrix operand order repaired and checked by reading, not compiled, not run on hardware
//
#include <hip/hip_runtime.h>
#include <hip/hip_bf16.h>

// ---------------------------------------------------------------------------
// BotDCGC Graph AutoEncoder for MI455X (gfx1250).
// Dominant cost: 4x dense 4096^3 GEMMs (B-power chain) + att@H GEMMs.
// GEMM: v_wmma_f32_16x16x32_f16, A-tiles fetched by the Tensor Data Mover
// (tensor_load_to_lds, double buffered, s_wait_tensorcnt), B-tiles staged
// transposed in LDS cooperatively.
// ---------------------------------------------------------------------------

#define USE_TDM 1   // set 0 to fall back to cooperative A staging

typedef _Float16 v16h __attribute__((ext_vector_type(16)));
typedef _Float16 v8h  __attribute__((ext_vector_type(8)));
typedef float    v8f  __attribute__((ext_vector_type(8)));
typedef unsigned int u32x4 __attribute__((ext_vector_type(4)));
typedef int          i32x8 __attribute__((ext_vector_type(8)));
typedef int          i32x4 __attribute__((ext_vector_type(4)));

#define NDIM 4096
#define NNL  (4096L * 4096L)
#define BM 128
#define BN 128
#define BK 32
#define LDT 40   // padded LDS stride in halves (32 + 8); 80B keeps 16B alignment

// ------------------------------ small kernels ------------------------------

__global__ __launch_bounds__(256) void k_zero_f32(float* __restrict__ p, long n) {
    long i = (long)blockIdx.x * 256 + threadIdx.x;
    if (i < n) p[i] = 0.0f;
}

__global__ __launch_bounds__(256) void k_scale_f32(float* __restrict__ p, long n, float s) {
    long i = (long)blockIdx.x * 256 + threadIdx.x;
    if (i < n) p[i] *= s;
}

__global__ __launch_bounds__(256) void k_cvt_f32_f16(const float* __restrict__ s,
                                                     _Float16* __restrict__ d, long n) {
    long i = (long)blockIdx.x * 256 + threadIdx.x;
    if (i < n) d[i] = (_Float16)s[i];
}

__global__ __launch_bounds__(256) void k_build_adj(const int* __restrict__ ei, int nE,
                                                   float* __restrict__ A) {
    int i = blockIdx.x * 256 + threadIdx.x;
    if (i >= nE) return;
    int e0 = ei[i];
    int e1 = ei[nE + i];
    if ((unsigned)e0 < NDIM && (unsigned)e1 < NDIM) {
        A[(size_t)e0 * NDIM + e1] = 1.0f;   // same-value races are benign
        A[(size_t)e1 * NDIM + e0] = 1.0f;
    }
}

// Row-normalize A -> B (f16 for WMMA) and seed M accumulator with B (f32).
__global__ __launch_bounds__(256) void k_row_norm(const float* __restrict__ A,
                                                  _Float16* __restrict__ B16,
                                                  float* __restrict__ Mout) {
    __shared__ float red[256];
    int i = blockIdx.x, t = threadIdx.x;
    float s = 0.0f;
    for (int j = t; j < NDIM; j += 256) s += A[(size_t)i * NDIM + j];
    red[t] = s; __syncthreads();
    for (int k = 128; k > 0; k >>= 1) { if (t < k) red[t] += red[t + k]; __syncthreads(); }
    float inv = 1.0f / fmaxf(red[0], 1.0f);
    for (int j = t; j < NDIM; j += 256) {
        float b = A[(size_t)i * NDIM + j] * inv;
        B16[(size_t)i * NDIM + j] = (_Float16)b;
        Mout[(size_t)i * NDIM + j] = b;
    }
}

// s1[i] = H[i,:] . a[0:Fo] ; s2[i] = H[i,:] . a[Fo:2Fo]
__global__ __launch_bounds__(256) void k_s1s2(const float* __restrict__ H,
                                              const float* __restrict__ a, int Fo,
                                              float* __restrict__ s1, float* __restrict__ s2) {
    __shared__ float r1[256], r2[256];
    int i = blockIdx.x, t = threadIdx.x;
    float p1 = 0.0f, p2 = 0.0f;
    for (int c = t; c < Fo; c += 256) {
        float h = H[(size_t)i * Fo + c];
        p1 += h * a[c];
        p2 += h * a[Fo + c];
    }
    r1[t] = p1; r2[t] = p2; __syncthreads();
    for (int k = 128; k > 0; k >>= 1) {
        if (t < k) { r1[t] += r1[t + k]; r2[t] += r2[t + k]; }
        __syncthreads();
    }
    if (t == 0) { s1[i] = r1[0]; s2[i] = r2[0]; }
}

// Row-wise masked softmax; rows with no neighbors get the identity row.
__global__ __launch_bounds__(256) void k_attention(const float* __restrict__ M,
                                                   const float* __restrict__ s1v,
                                                   const float* __restrict__ s2v,
                                                   _Float16* __restrict__ att, int n) {
    __shared__ float red[256];
    int i = blockIdx.x, t = threadIdx.x;
    const float NEG = -3.0e38f;
    float s1i = s1v[i];

    float lmax = NEG;
    for (int j = t; j < n; j += 256) {
        float m = M[(size_t)i * n + j];
        if (m > 0.0f) {
            float e = s1i + s2v[j];
            e = (e >= 0.0f) ? e : 0.2f * e;
            lmax = fmaxf(lmax, m * e);
        }
    }
    red[t] = lmax; __syncthreads();
    for (int k = 128; k > 0; k >>= 1) { if (t < k) red[t] = fmaxf(red[t], red[t + k]); __syncthreads(); }
    float rmax = red[0]; __syncthreads();

    if (rmax < -1.0e37f) {   // no neighbors: one-hot self row
        for (int j = t; j < n; j += 256)
            att[(size_t)i * n + j] = (_Float16)((j == i) ? 1.0f : 0.0f);
        return;
    }
    float lsum = 0.0f;
    for (int j = t; j < n; j += 256) {
        float m = M[(size_t)i * n + j];
        if (m > 0.0f) {
            float e = s1i + s2v[j];
            e = (e >= 0.0f) ? e : 0.2f * e;
            lsum += __expf(m * e - rmax);
        }
    }
    red[t] = lsum; __syncthreads();
    for (int k = 128; k > 0; k >>= 1) { if (t < k) red[t] += red[t + k]; __syncthreads(); }
    float inv = 1.0f / red[0];
    for (int j = t; j < n; j += 256) {
        float m = M[(size_t)i * n + j];
        float v = 0.0f;
        if (m > 0.0f) {
            float e = s1i + s2v[j];
            e = (e >= 0.0f) ? e : 0.2f * e;
            v = __expf(m * e - rmax) * inv;
        }
        att[(size_t)i * n + j] = (_Float16)v;
    }
}

__global__ __launch_bounds__(256) void k_transpose16(const _Float16* __restrict__ Z,
                                                     _Float16* __restrict__ Zt,
                                                     int rows, int cols) {
    int i = blockIdx.x * 256 + threadIdx.x;
    if (i < rows * cols) {
        int r = i / cols, c = i % cols;
        Zt[(size_t)c * rows + r] = Z[(size_t)r * cols + c];
    }
}

// ------------------------------- WMMA GEMM ---------------------------------

static __device__ inline v8f wmma16(v16h a, v16h b, v8f c) {
    return __builtin_amdgcn_wmma_f32_16x16x32_f16(false, a, false, b, (short)0, c,
                                                  false, false);
}

// A fragment (16x32, 16-bit) per ISA layout: lane = 16*kh + m;
// v16h[0..7] = row m, K = kh*8 + 0..7 ; v16h[8..15] = row m, K = 16+kh*8+0..7
static __device__ inline v16h load_frag_a(const _Float16* base, int stride) {
    int lane = threadIdx.x & 31;
    int m = lane & 15, kh = lane >> 4;
    const _Float16* p = base + m * stride + kh * 8;
    v8h lo = *(const v8h*)(p);
    v8h hi = *(const v8h*)(p + 16);
    return __builtin_shufflevector(lo, hi, 0,1,2,3,4,5,6,7,8,9,10,11,12,13,14,15);
}

// B fragment (32x16): lane = 16*kh + n; v16h[0..7] = K kh*8+0..7,
// v16h[8..15] = K 16+kh*8+0..7. Tile stored transposed ([n][k]) in LDS.
static __device__ inline v16h load_frag_b(const _Float16* baseT, int stride) {
    int lane = threadIdx.x & 31;
    int n = lane & 15, kh = lane >> 4;
    const _Float16* p = baseT + n * stride + kh * 8;
    v8h lo = *(const v8h*)(p);
    v8h hi = *(const v8h*)(p + 16);
    return __builtin_shufflevector(lo, hi, 0,1,2,3,4,5,6,7,8,9,10,11,12,13,14,15);
}

static __device__ inline float dsigmoid(float x) { return 1.0f / (1.0f + __expf(-x)); }

#if USE_TDM
// Issue a TDM load of a BK x BM (32 x 128) f16 tile of A into LDS.
// D# per CDNA5 ISA sec.8. Pad fields produce the padded
// LDS row stride: 16 DWORDs data (32 halves) then 4 DWORDs pad -> LDT = 40.
// NOTE: this toolchain's builtin takes 6 args (g0,g1,g2,g3,extra_g,cpol).
static __device__ inline void tdm_load_a(const _Float16* A, int lda, int rowBase,
                                         int k0, int Mdim, int Kdim,
                                         _Float16* ldsDst) {
    unsigned long long ga =
        (unsigned long long)(size_t)(A + (size_t)rowBase * lda + k0);
    unsigned int ldsOff = (unsigned int)(size_t)ldsDst;  // low 32b = LDS offset

    u32x4 g0;
    g0[0] = 1u;                                            // count=1, user mode
    g0[1] = ldsOff;                                        // lds_addr
    g0[2] = (unsigned int)(ga & 0xffffffffull);            // global_addr[31:0]
    g0[3] = (unsigned int)((ga >> 32) & 0x01ffffffull)     // global_addr[56:32]
          | (2u << 30);                                    // type = 2 (image)

    unsigned int td0 = (unsigned int)(Kdim - k0);          // remaining extent, dim0
    unsigned int td1 = (unsigned int)(Mdim - rowBase);     // remaining extent, dim1
    i32x8 g1;
    g1[0] = (int)((1u << 16)      // data_size = 2 bytes
                | (1u << 20)      // pad_enable
                | (3u << 22)      // pad_interval: 16 DWORDs
                | (3u << 25));    // pad_amount: 4 DWORDs
    g1[1] = (int)((td0 & 0xffffu) << 16);                  // tensor_dim0[15:0]
    g1[2] = (int)((td0 >> 16) | ((td1 & 0xffffu) << 16));  // td0 hi | td1 lo
    g1[3] = (int)((td1 >> 16) | ((unsigned)BK << 16));     // td1 hi | tile_dim0
    g1[4] = (int)BM;                                       // tile_dim1 (tile_dim2=0)
    g1[5] = (int)(unsigned)lda;                            // tensor_dim0_stride lo
    g1[6] = 0;                                             // stride hi | dim1_stride lo
    g1[7] = 0;
    i32x4 z4 = {0, 0, 0, 0};
    i32x8 z8 = {0, 0, 0, 0, 0, 0, 0, 0};
    __builtin_amdgcn_tensor_load_to_lds(g0, g1, z4, z4, z8, 0);
}
#endif

// C(MdxNd,f32) = A(MdxKd,f16) * B(KdxNd,f16); optional f16 copy, optional
// accumulate into Acc, optional sigmoid. 8 wave32 waves, each owns a 64x32
// tile -> 8x v_wmma_f32_16x16x32_f16 per BK step. A tiles via TDM (double
// buffered); B tiles staged transposed cooperatively.
__global__ __launch_bounds__(256) void k_gemm_wmma(
    const _Float16* __restrict__ A, int lda,
    const _Float16* __restrict__ B, int ldb,
    float* __restrict__ Cf32, _Float16* __restrict__ Cf16, float* __restrict__ Acc,
    int ldc, int Mdim, int Ndim, int Kdim, int epilogue) {

    __shared__ _Float16 sA[2][BM][LDT];   // A tiles, double buffered (TDM dest)
    __shared__ _Float16 sBt[BN][LDT];     // B tile, transposed ([n][k])

    const int rowBase = blockIdx.y * BM;
    const int colBase = blockIdx.x * BN;
    const int waveId = threadIdx.x >> 5;
    const int rw = (waveId >> 2) * 64;    // wave row origin (2 wave-rows)
    const int cw = (waveId & 3) * 32;     // wave col origin (4 wave-cols)

    v8f acc[4][2];
    for (int a0 = 0; a0 < 4; ++a0)
        for (int b0 = 0; b0 < 2; ++b0)
            acc[a0][b0] = (v8f)0.0f;

    const bool tdmWave = (waveId == 0);
    int buf = 0;
#if USE_TDM
    if (tdmWave) tdm_load_a(A, lda, rowBase, 0, Mdim, Kdim, &sA[0][0][0]);
#endif

    for (int k0 = 0; k0 < Kdim; k0 += BK) {
        const bool hasNext = (k0 + BK) < Kdim;
#if USE_TDM
        if (tdmWave && hasNext)
            tdm_load_a(A, lda, rowBase, k0 + BK, Mdim, Kdim, &sA[buf ^ 1][0][0]);
#else
        {   // cooperative A staging (fallback): 128x32 halves, 16/thread
            int r = threadIdx.x >> 1;
            int cseg = (threadIdx.x & 1) * 16;
            int grow = rowBase + r, gc0 = k0 + cseg;
            if (grow < Mdim && gc0 + 16 <= Kdim) {
                const v8h* gp = (const v8h*)(A + (size_t)grow * lda + gc0);
                *(v8h*)&sA[buf][r][cseg]     = gp[0];
                *(v8h*)&sA[buf][r][cseg + 8] = gp[1];
            } else {
                int rr = grow < Mdim ? grow : Mdim - 1;
                for (int x = 0; x < 16; ++x) {
                    int gc = gc0 + x;
                    int cc = gc < Kdim ? gc : Kdim - 1;
                    _Float16 v = A[(size_t)rr * lda + cc];
                    sA[buf][r][cseg + x] =
                        (grow < Mdim && gc < Kdim) ? v : (_Float16)0;
                }
            }
        }
#endif
        // ---- stage B tile transposed: 32x128 halves, 16 per thread ----
        {
            int r  = threadIdx.x >> 3;          // k within tile (0..31)
            int c0 = (threadIdx.x & 7) * 16;    // n within tile (0..112)
            int gk = k0 + r;
            int gc0 = colBase + c0;
            _Float16 tmp[16];
            if (gk < Kdim && gc0 + 16 <= Ndim) {
                const v8h* gp = (const v8h*)(B + (size_t)gk * ldb + gc0);
                *(v8h*)&tmp[0] = gp[0];
                *(v8h*)&tmp[8] = gp[1];
                if (hasNext)   // speculative prefetch of next K tile
                    __builtin_prefetch(B + (size_t)(gk + BK) * ldb + gc0, 0, 1);
            } else {
                // clamped unconditional loads + select: no exec-mask branching
                int rr = gk < Kdim ? gk : Kdim - 1;
                for (int x = 0; x < 16; ++x) {
                    int gc = gc0 + x;
                    int cc = gc < Ndim ? gc : Ndim - 1;
                    _Float16 v = B[(size_t)rr * ldb + cc];
                    tmp[x] = (gk < Kdim && gc < Ndim) ? v : (_Float16)0;
                }
            }
            for (int x = 0; x < 16; ++x) sBt[c0 + x][r] = tmp[x];
        }
#if USE_TDM
        if (tdmWave) {
            if (hasNext) __builtin_amdgcn_s_wait_tensorcnt(1);
            else         __builtin_amdgcn_s_wait_tensorcnt(0);
        }
#endif
        __syncthreads();

        v16h af0 = load_frag_a(&sA[buf][rw +  0][0], LDT);
        v16h af1 = load_frag_a(&sA[buf][rw + 16][0], LDT);
        v16h af2 = load_frag_a(&sA[buf][rw + 32][0], LDT);
        v16h af3 = load_frag_a(&sA[buf][rw + 48][0], LDT);
        v16h b0  = load_frag_b(&sBt[cw +  0][0], LDT);
        v16h b1  = load_frag_b(&sBt[cw + 16][0], LDT);

        acc[0][0] = wmma16(af0, b0, acc[0][0]);
        acc[0][1] = wmma16(af0, b1, acc[0][1]);
        acc[1][0] = wmma16(af1, b0, acc[1][0]);
        acc[1][1] = wmma16(af1, b1, acc[1][1]);
        acc[2][0] = wmma16(af2, b0, acc[2][0]);
        acc[2][1] = wmma16(af2, b1, acc[2][1]);
        acc[3][0] = wmma16(af3, b0, acc[3][0]);
        acc[3][1] = wmma16(af3, b1, acc[3][1]);
        __syncthreads();
        buf ^= 1;
    }

    // ---- epilogue: C/D layout -> col = lane&15, row = r + 8*(lane>>4) ----
    int lane = threadIdx.x & 31;
    int cIn = lane & 15, rHalf = lane >> 4;
    for (int mi = 0; mi < 4; ++mi)
        for (int ni = 0; ni < 2; ++ni)
            for (int r = 0; r < 8; ++r) {
                int grow = rowBase + rw + mi * 16 + r + 8 * rHalf;
                int gcol = colBase + cw + ni * 16 + cIn;
                if (grow < Mdim && gcol < Ndim) {
                    float v = acc[mi][ni][r];
                    size_t idx = (size_t)grow * ldc + gcol;
                    if (Acc) Acc[idx] += v;
                    if (epilogue == 1) v = dsigmoid(v);
                    if (Cf32) Cf32[idx] = v;
                    if (Cf16) Cf16[idx] = (_Float16)v;
                }
            }
}

// ------------------------------- launcher ----------------------------------

extern "C" void kernel_launch(void* const* d_in, const int* in_sizes, int n_in,
                              void* d_out, int out_size, void* d_ws, size_t ws_size,
                              hipStream_t stream) {
    (void)n_in; (void)out_size; (void)ws_size;

    const float* X  = (const float*)d_in[0];
    const int*   ei = (const int*)d_in[1];
    const float* W[3]  = {(const float*)d_in[2], (const float*)d_in[4], (const float*)d_in[6]};
    const float* av[3] = {(const float*)d_in[3], (const float*)d_in[5], (const float*)d_in[7]};
    const int nE = in_sizes[1] / 2;

    // output layout: Z (4096x16), A_hat (NxN), A (NxN), M (NxN)
    float* outZ    = (float*)d_out;
    float* outAhat = outZ + (size_t)NDIM * 16;
    float* outA    = outAhat + NNL;
    float* outM    = outA + NNL;

    // workspace carve-up (~114 MB)
    char* w = (char*)d_ws; size_t off = 0;
    auto alloc = [&](size_t b) -> char* {
        char* p = w + off; off += (b + 255) & ~(size_t)255; return p;
    };
    _Float16* B16  = (_Float16*)alloc(NNL * 2);
    _Float16* Pa   = (_Float16*)alloc(NNL * 2);
    _Float16* Pb   = (_Float16*)alloc(NNL * 2);
    _Float16* X16  = (_Float16*)alloc((size_t)NDIM * 256 * 2);
    _Float16* W16  = (_Float16*)alloc(256 * 256 * 2);
    float*    H32  = (float*)   alloc((size_t)NDIM * 256 * 4);
    _Float16* H16  = (_Float16*)alloc((size_t)NDIM * 256 * 2);
    _Float16* Z16a = (_Float16*)alloc((size_t)NDIM * 256 * 2);
    _Float16* Z16b = (_Float16*)alloc((size_t)NDIM * 256 * 2);
    _Float16* Zt16 = (_Float16*)alloc((size_t)16 * NDIM * 2);
    float*    s1   = (float*)   alloc(NDIM * 4);
    float*    s2   = (float*)   alloc(NDIM * 4);

    auto gemm = [&](const _Float16* A_, int lda, const _Float16* B_, int ldb,
                    float* C32, _Float16* C16, float* Acc, int ldc,
                    int Md, int Nd, int Kd, int ep) {
        dim3 g((Nd + BN - 1) / BN, (Md + BM - 1) / BM);
        k_gemm_wmma<<<g, dim3(256), 0, stream>>>(A_, lda, B_, ldb, C32, C16, Acc,
                                                 ldc, Md, Nd, Kd, ep);
    };

    // 1) adjacency A
    k_zero_f32<<<(int)((NNL + 255) / 256), 256, 0, stream>>>(outA, NNL);
    k_build_adj<<<(nE + 255) / 256, 256, 0, stream>>>(ei, nE, outA);

    // 2) B = row-normalized A (f16); M accumulator seeded with B
    k_row_norm<<<NDIM, 256, 0, stream>>>(outA, B16, outM);

    // 3) power chain: M += B^2 + B^3 + B^4 + B^5 (accumulated in GEMM epilogue)
    {
        const _Float16* prev = B16;
        _Float16* nxt = Pa;
        for (int s = 0; s < 4; ++s) {
            gemm(prev, NDIM, B16, NDIM, nullptr, nxt, outM, NDIM,
                 NDIM, NDIM, NDIM, 0);
            prev = nxt;
            nxt = (nxt == Pa) ? Pb : Pa;
        }
    }
    k_scale_f32<<<(int)((NNL + 255) / 256), 256, 0, stream>>>(outM, NNL, 0.2f);

    // 4) three GAT layers
    k_cvt_f32_f16<<<(NDIM * 256 + 255) / 256, 256, 0, stream>>>(X, X16, (long)NDIM * 256);
    const _Float16* cur = X16;
    int Fin = 256;
    const int Fouts[3] = {256, 256, 16};
    _Float16* att = Pa;           // free after power chain
    _Float16* Zp[2] = {Z16a, Z16b};
    for (int l = 0; l < 3; ++l) {
        int Fo = Fouts[l];
        k_cvt_f32_f16<<<(Fin * Fo + 255) / 256, 256, 0, stream>>>(W[l], W16, (long)Fin * Fo);
        // H = Xl @ W
        gemm(cur, Fin, W16, Fo, H32, H16, nullptr, Fo, NDIM, Fo, Fin, 0);
        // attention scores
        k_s1s2<<<NDIM, 256, 0, stream>>>(H32, av[l], Fo, s1, s2);
        k_attention<<<NDIM, 256, 0, stream>>>(outM, s1, s2, att, NDIM);
        // Z = sigmoid(att @ H); last layer's f32 result goes to d_out
        float* c32 = (l == 2) ? outZ : nullptr;
        gemm(att, NDIM, H16, Fo, c32, Zp[l & 1], nullptr, Fo, NDIM, NDIM, NDIM, 1);
        cur = Zp[l & 1];
        Fin = Fo;
    }

    // 5) A_hat = sigmoid(Z @ Z^T) via WMMA (K=16, zero-padded to BK)
    k_transpose16<<<(NDIM * 16 + 255) / 256, 256, 0, stream>>>(cur, Zt16, NDIM, 16);
    gemm(cur, 16, Zt16, NDIM, outAhat, nullptr, nullptr, NDIM, NDIM, NDIM, 16, 1);
}
